// SynthesisLayer_78357383348529
// MI455X (gfx1250) — compile-verified, matrix operand order checked by reading, not run on hardware
//
#include <hip/hip_runtime.h>
#include <hip/hip_bf16.h>

#define B_    16
#define CIN_  512
#define COUT_ 512
#define RES_  64
#define KTOT_ (9 * 512)    // 4608 = GEMM K dimension
#define NK_   144          // 9 taps * 16 cin-blocks of 32

typedef __attribute__((ext_vector_type(16))) __bf16 v16bf;
typedef __attribute__((ext_vector_type(8)))  __bf16 v8bf;
typedef __attribute__((ext_vector_type(8)))  float  v8f;
typedef int v4i __attribute__((vector_size(16)));   // builtin's pointee type

#define AS1 __attribute__((address_space(1)))
#define AS3 __attribute__((address_space(3)))

#if __has_builtin(__builtin_amdgcn_global_load_async_to_lds_b128) && \
    __has_builtin(__builtin_amdgcn_s_wait_asynccnt)
#define USE_ASYNC 1
#else
#define USE_ASYNC 0
#endif

union FragBF {
    v16bf v;
    v8bf  h[2];
};

// 16-byte global -> LDS copy: async-to-LDS on CDNA5, sync fallback otherwise.
__device__ __forceinline__ void cp16(void* lds, const void* g) {
#if USE_ASYNC
    __builtin_amdgcn_global_load_async_to_lds_b128(
        (AS1 v4i*)g, (AS3 v4i*)lds, 0, 0);
#else
    *(v8bf*)lds = *(const v8bf*)g;
#endif
}

// ---------------------------------------------------------------------------
// styles[b][cin] = dot(w[b,:], affine_w[cin,:]) / sqrt(512) + affine_b[cin]
// Also zeroes the 256B OOB pad region used by the async halo loads.
// ---------------------------------------------------------------------------
__global__ void styles_kernel(const float* __restrict__ w,
                              const float* __restrict__ aw,
                              const float* __restrict__ ab,
                              float* __restrict__ styles,
                              float* __restrict__ zpad) {
    int idx = blockIdx.x * blockDim.x + threadIdx.x;   // b*CIN + cin
    if (blockIdx.x == 0 && threadIdx.x < 64) zpad[threadIdx.x] = 0.f;
    if (idx >= B_ * CIN_) return;
    int b = idx >> 9, cin = idx & (CIN_ - 1);
    const float* wr = w  + (size_t)b   * CIN_;
    const float* ar = aw + (size_t)cin * CIN_;
    float s = 0.f;
    for (int k = 0; k < CIN_; k += 4) {
        s = fmaf(wr[k + 0], ar[k + 0], s);
        s = fmaf(wr[k + 1], ar[k + 1], s);
        s = fmaf(wr[k + 2], ar[k + 2], s);
        s = fmaf(wr[k + 3], ar[k + 3], s);
    }
    styles[idx] = s * 0.04419417382415922f + ab[cin];  // 1/sqrt(512)
}

// ---------------------------------------------------------------------------
// wsq[cout][cin] = sum over 3x3 taps of weight^2
// ---------------------------------------------------------------------------
__global__ void wsq_kernel(const float* __restrict__ weight,
                           float* __restrict__ wsq) {
    int idx = blockIdx.x * blockDim.x + threadIdx.x;   // cout*CIN + cin
    if (idx >= COUT_ * CIN_) return;
    const float* p = weight + (size_t)idx * 9;
    float s = 0.f;
#pragma unroll
    for (int j = 0; j < 9; ++j) s = fmaf(p[j], p[j], s);
    wsq[idx] = s;
}

// ---------------------------------------------------------------------------
// dcoef[b][cout] = rsqrt( sum_cin wsq[cout][cin]*styles[b][cin]^2 + 1e-8 )
// ---------------------------------------------------------------------------
__global__ void dcoef_kernel(const float* __restrict__ wsq,
                             const float* __restrict__ styles,
                             float* __restrict__ dcoef) {
    int idx = blockIdx.x * blockDim.x + threadIdx.x;   // b*COUT + cout
    if (idx >= B_ * COUT_) return;
    int b = idx >> 9, cout = idx & (COUT_ - 1);
    const float* wr = wsq    + (size_t)cout * CIN_;
    const float* sr = styles + (size_t)b    * CIN_;
    float s = 0.f;
    for (int c = 0; c < CIN_; ++c) {
        float st = sr[c];
        s = fmaf(wr[c], st * st, s);
    }
    dcoef[idx] = rsqrtf(s + 1e-8f);
}

// ---------------------------------------------------------------------------
// wt[cout][r][cin] (bf16)  <-  weight[cout][cin][ky][kx] (f32), r = ky*3+kx
// ---------------------------------------------------------------------------
__global__ void wt_kernel(const float* __restrict__ weight,
                          __bf16* __restrict__ wt) {
    int idx = blockIdx.x * blockDim.x + threadIdx.x;   // cout*4608 + r*512 + cin
    if (idx >= COUT_ * KTOT_) return;
    int cout = idx / KTOT_;
    int rem  = idx - cout * KTOT_;
    int r    = rem >> 9;
    int cin  = rem & (CIN_ - 1);
    wt[idx] = (__bf16)weight[((size_t)cout * CIN_ + cin) * 9 + r];
}

// ---------------------------------------------------------------------------
// xmod NHWC (bf16): xmod[b][h][w][cin] = x[b][cin][h][w] * styles[b][cin]
// cin-contiguous so conv B-tiles stage with single b128 loads.
// ---------------------------------------------------------------------------
__global__ void xmod_kernel(const float* __restrict__ x,
                            const float* __restrict__ styles,
                            __bf16* __restrict__ xmod) {
    int idx = blockIdx.x * blockDim.x + threadIdx.x;   // (b,h,w,cin) cin fastest
    if (idx >= B_ * CIN_ * RES_ * RES_) return;
    int cin = idx & (CIN_ - 1);
    int t   = idx >> 9;            // b*4096 + pix
    int pix = t & 4095;
    int b   = t >> 12;
    float s = styles[(b << 9) | cin];
    xmod[idx] = (__bf16)(x[(((size_t)b * CIN_ + cin) << 12) | pix] * s);
}

// ---------------------------------------------------------------------------
// Stage one K-step's A (weights) and B (patch) tiles into LDS buffer `buf`.
// 3 x 16B transfers per thread; uniform async-op count (OOB -> zero pad).
// ---------------------------------------------------------------------------
__device__ __forceinline__ void stage(int ks, __bf16* asb, __bf16* bsb,
                                      const __bf16* wbase, const __bf16* xb,
                                      const __bf16* zpad, int h,
                                      int arow, int acol, int bw, int bk8) {
    const int r  = ks >> 4;
    const int cb = (ks & 15) << 5;
    const int ky = r / 3 - 1;
    const int kx = r - (r / 3) * 3 - 1;
    // A: 128 cout-rows x 32 cin, contiguous in wt layout
    const __bf16* srcA = wbase + (size_t)arow * KTOT_ + r * CIN_ + cb + acol;
    cp16(asb + arow * 32 + acol,     srcA);
    cp16(asb + arow * 32 + acol + 8, srcA + 8);
    // B: pixel-major [w][cin] tile from NHWC activations (halo -> zero pad)
    const int hs  = h + ky;
    const int wsx = bw + kx;
    const bool inb = (hs >= 0) && (hs < RES_) && (wsx >= 0) && (wsx < RES_);
    const __bf16* srcB = inb
        ? xb + ((size_t)hs * RES_ + wsx) * CIN_ + cb + bk8
        : zpad;
    cp16(bsb + bw * 32 + bk8, srcB);
}

// ---------------------------------------------------------------------------
// Implicit-GEMM 3x3 conv, batch-shared bf16 weights, double-buffered async
// LDS pipeline, fused demod/noise/bias/lrelu/gain epilogue.
// Block tile: M=128 (cout) x N=64 (one image row). 8 waves, 4 subtiles/wave.
// ---------------------------------------------------------------------------
__global__ __launch_bounds__(256) void conv_wmma_kernel(
    const __bf16* __restrict__ xmod,   // [B][64][64][CIN] NHWC
    const __bf16* __restrict__ wt,     // [COUT][9][CIN]
    const __bf16* __restrict__ zpad,   // 256B of zeros
    const float*  __restrict__ dcoef,  // [B][COUT]
    const float*  __restrict__ bias,   // [COUT]
    const float*  __restrict__ noise,  // [64][64]
    const float*  __restrict__ nstr,   // scalar
    float* __restrict__ out)           // [B][COUT][64][64]
{
    __shared__ __align__(16) __bf16 As[2][128][32];  // 2 x 8KB
    __shared__ __align__(16) __bf16 Bs[2][64][32];   // 2 x 4KB

    const int tid  = threadIdx.x;
    const int lane = tid & 31;
    const int wave = tid >> 5;

    const int bid = blockIdx.x;
    const int h   = bid & 63;          // output row (N tile)
    const int mt  = (bid >> 6) & 3;    // cout tile
    const int b   = bid >> 8;          // batch
    const int m0  = mt * 128;

    const float fstr = nstr[0];

    const __bf16* xb    = xmod + (size_t)b * RES_ * RES_ * CIN_;
    const __bf16* wbase = wt   + (size_t)m0 * KTOT_;

    const int arow = tid >> 1;          // A staging: 32B per thread
    const int acol = (tid & 1) * 16;
    const int bw   = tid & 63;          // B staging: 16B per thread
    const int bk8  = (tid >> 6) * 8;

    v8f acc[4];
#pragma unroll
    for (int t = 0; t < 4; ++t)
#pragma unroll
        for (int i = 0; i < 8; ++i) acc[t][i] = 0.f;

    const int g  = lane >> 4;          // K-group per ISA A/B fragment layout
    const int lr = lane & 15;          // row (A) / col (B) within subtile

    // prologue: fill buffer 0
    stage(0, &As[0][0][0], &Bs[0][0][0], wbase, xb, zpad, h, arow, acol, bw, bk8);

    for (int ks = 0; ks < NK_; ++ks) {
        const int buf = ks & 1;
        if (ks + 1 < NK_) {
            stage(ks + 1, &As[buf ^ 1][0][0], &Bs[buf ^ 1][0][0],
                  wbase, xb, zpad, h, arow, acol, bw, bk8);
#if USE_ASYNC
            __builtin_amdgcn_s_wait_asynccnt(3);  // only next-tile ops outstanding
#endif
        } else {
#if USE_ASYNC
            __builtin_amdgcn_s_wait_asynccnt(0);
#endif
        }
        __syncthreads();   // all waves' tile-ks async writes visible

        FragBF fa;
        fa.h[0] = *(const v8bf*)(&As[buf][wave * 16 + lr][g * 8]);
        fa.h[1] = *(const v8bf*)(&As[buf][wave * 16 + lr][16 + g * 8]);
#pragma unroll
        for (int nt = 0; nt < 4; ++nt) {
            FragBF fb;
            fb.h[0] = *(const v8bf*)(&Bs[buf][nt * 16 + lr][g * 8]);
            fb.h[1] = *(const v8bf*)(&Bs[buf][nt * 16 + lr][16 + g * 8]);
            acc[nt] = __builtin_amdgcn_wmma_f32_16x16x32_bf16(
                false, fa.v, false, fb.v, (short)0, acc[nt], false, false);
        }
        __syncthreads();   // all reads of buf done before it is refilled
    }

    // ---- epilogue: demod scale, noise, bias, leaky-relu, sqrt(2) gain ----
#pragma unroll
    for (int nt = 0; nt < 4; ++nt) {
        const int wpix = nt * 16 + lr;
        const float nz = noise[h * RES_ + wpix] * fstr;
#pragma unroll
        for (int i = 0; i < 8; ++i) {
            const int m = m0 + wave * 16 + g * 8 + i;   // cout
            float v = acc[nt][i];
            v *= dcoef[b * COUT_ + m];
            v += nz;
            v += bias[m];
            v = (v > 0.f) ? v : v * 0.2f;
            v *= 1.4142135623730951f;
            out[(((size_t)b * COUT_ + m) * RES_ + h) * RES_ + wpix] = v;
        }
    }
}

// ---------------------------------------------------------------------------
extern "C" void kernel_launch(void* const* d_in, const int* in_sizes, int n_in,
                              void* d_out, int out_size, void* d_ws, size_t ws_size,
                              hipStream_t stream) {
    const float* x       = (const float*)d_in[0];
    const float* w       = (const float*)d_in[1];
    const float* weight  = (const float*)d_in[2];
    const float* aff_w   = (const float*)d_in[3];
    const float* aff_b   = (const float*)d_in[4];
    const float* bias    = (const float*)d_in[5];
    const float* noise   = (const float*)d_in[6];
    const float* nstr    = (const float*)d_in[7];
    float* out = (float*)d_out;

    // workspace layout (bytes), 256-aligned regions
    char* ws = (char*)d_ws;
    float*  zpad   = (float*)(ws);                                   // 256 B
    float*  styles = (float*)(ws + 256);                             // 32 KB
    float*  dcoef  = (float*)(ws + 256 + 32768);                     // 32 KB
    float*  wsq    = (float*)(ws + 256 + 65536);                     // 1 MB
    __bf16* wt     = (__bf16*)(ws + 256 + 65536 + 1048576);          // 4.5 MB
    __bf16* xmod   = (__bf16*)(ws + 256 + 65536 + 1048576
                               + (size_t)COUT_ * KTOT_ * 2);         // 64 MB

    styles_kernel<<<(B_ * CIN_ + 255) / 256, 256, 0, stream>>>(w, aff_w, aff_b,
                                                               styles, zpad);
    wsq_kernel<<<(COUT_ * CIN_ + 255) / 256, 256, 0, stream>>>(weight, wsq);
    dcoef_kernel<<<(B_ * COUT_ + 255) / 256, 256, 0, stream>>>(wsq, styles, dcoef);
    wt_kernel<<<(COUT_ * KTOT_ + 255) / 256, 256, 0, stream>>>(weight, wt);
    xmod_kernel<<<(B_ * CIN_ * RES_ * RES_ + 255) / 256, 256, 0, stream>>>(x, styles, xmod);

    // 16 batches x 4 cout-tiles x 64 rows = 4096 workgroups
    conv_wmma_kernel<<<B_ * 4 * RES_, 256, 0, stream>>>(
        xmod, wt, (const __bf16*)zpad, dcoef, bias, noise, nstr, out);
}
